// GMPool_1_67181878444696
// MI455X (gfx1250) — matrix operand order repaired and checked by reading, compiled, not run on hardware
//
#include <hip/hip_runtime.h>
#include <hip/hip_bf16.h>

typedef __attribute__((ext_vector_type(16))) _Float16 v16h;
typedef __attribute__((ext_vector_type(8)))  _Float16 v8h;
typedef __attribute__((ext_vector_type(8)))  float    v8f;

constexpr int NB    = 16;    // batch
constexpr int NT    = 176;   // time / groups
constexpr int NROI  = 116;   // ROI
constexpr int NHID  = 200;   // hidden
constexpr int NMAXG = 176;
constexpr int NPAIR = 6670;  // 116*115/2
constexpr int NHT   = 13;    // ceil(200/16) hid tiles
constexpr int NHP   = NHT * 16;  // 208 padded hidden
constexpr int NSWEEP = 12;   // Jacobi sweeps
constexpr float TAU_INV = 1e5f;

__device__ inline v16h cat8(v8h lo, v8h hi) {
  return __builtin_shufflevector(lo, hi, 0, 1, 2, 3, 4, 5, 6, 7,
                                 8, 9, 10, 11, 12, 13, 14, 15);
}

// ---------------------------------------------------------------------------
// Kernel 0: one-shot padding/conversion of the MLP weights.
// W1h: [208][128] f16 zero-padded (N-tile-major, K padded 116->128) so the
// clf hot loop does unconditional aligned 16B vector loads. b1p/W2p padded.
// ---------------------------------------------------------------------------
__global__ __launch_bounds__(256) void prep_kernel(
    const float* __restrict__ W1, const float* __restrict__ b1,
    const float* __restrict__ W2, _Float16* __restrict__ W1h,
    float* __restrict__ b1p, float* __restrict__ W2p) {
  const int idx = blockIdx.x * 256 + threadIdx.x;
  if (idx < NHP * 128) {
    const int n = idx >> 7, k = idx & 127;
    const float v = (n < NHID && k < NROI) ? W1[(size_t)n * NROI + k] : 0.f;
    W1h[idx] = (_Float16)v;
  }
  if (idx < NHP) {
    b1p[idx] = (idx < NHID) ? b1[idx] : 0.f;
    W2p[idx] = (idx < NHID) ? W2[idx] : 0.f;
  }
}

// ---------------------------------------------------------------------------
// Kernel 1: fused distance -> MLP -> sigmoid -> grouping_M.
// One wave (32 threads) per (b, t, 16-wide s-tile). M = 16 (t,s) pairs,
// K = ROI padded to 128 (4 WMMA k-steps of 32), N loops over 13 hid tiles.
// ---------------------------------------------------------------------------
__global__ __launch_bounds__(32) void clf_kernel(
    const float* __restrict__ X, const _Float16* __restrict__ W1h,
    const float* __restrict__ b1p, const float* __restrict__ W2p,
    const float* __restrict__ b2, float* __restrict__ outGM,
    float* __restrict__ wsA) {
  const int s0   = blockIdx.x * 16;
  const int t    = blockIdx.y;
  const int b    = blockIdx.z;
  const int lane = threadIdx.x;

  __shared__ __align__(16) _Float16 dist[16 * 128];  // 16 pairs x 128 K, f16

  const float* xb = X + (size_t)b * NT * NROI;
  const float* xt = xb + (size_t)t * NROI;

  // Build the distance tile: dist[m][k] = sqrt((X[t,k]-X[s0+m,k])^2 + 1e-9)
  for (int idx = lane; idx < 16 * 128; idx += 32) {
    const int m = idx >> 7, k = idx & 127;
    float v = 0.f;
    if (k < NROI) {
      const float d = xt[k] - xb[(size_t)(s0 + m) * NROI + k];
      v = sqrtf(d * d + 1e-9f);
    }
    dist[idx] = (_Float16)v;
  }
  __syncthreads();

  const int nlo = lane & 15;   // column / row-within-half index
  const int grp = lane >> 4;   // lane half
  const float b2v = b2[0];

  float clfacc[8];
#pragma unroll
  for (int j = 0; j < 8; ++j) clfacc[j] = 0.f;

  for (int ntile = 0; ntile < NHT; ++ntile) {
    const int hid = ntile * 16 + nlo;
    const float b1v = b1p[hid];
    const float w2v = W2p[hid];
    const _Float16* wrow = W1h + (size_t)hid * 128;
    v8f c = {};
#pragma unroll
    for (int ks = 0; ks < 4; ++ks) {
      const int kb = ks * 32;
      // A fragment (16x32 f16): lane holds row M=nlo;
      // half i -> K = kb + grp*16 + i (16 contiguous halves).
      const _Float16* arow = dist + nlo * 128 + kb + grp * 16;
      const v16h a = cat8(*(const v8h*)arow, *(const v8h*)(arow + 8));
      // B fragment (32x16 f16): B[k][n] = W1[hid=n][k]; lane holds N=nlo,
      // half i -> K = kb + grp*16 + i  (16 contiguous halves, 32B aligned).
      const _Float16* brow = wrow + kb + grp * 16;
      const v16h bm = cat8(*(const v8h*)brow, *(const v8h*)(brow + 8));
      c = __builtin_amdgcn_wmma_f32_16x16x32_f16(false, a, false, bm,
                                                 (short)0, c, false, false);
    }
    // C layout: c[j] = element (M = grp*8 + j, N = nlo). Apply +b1, ReLU, *W2.
#pragma unroll
    for (int j = 0; j < 8; ++j) {
      float h = c[j] + b1v;
      h = h > 0.f ? h : 0.f;
      clfacc[j] += h * w2v;
    }
  }

  // Reduce over the 16 hid lanes sharing the same M rows (xor 1,2,4,8).
#pragma unroll
  for (int j = 0; j < 8; ++j) {
    float v = clfacc[j];
    for (int m = 1; m < 16; m <<= 1) v += __shfl_xor(v, m, 32);
    clfacc[j] = v;
  }

  if (nlo == 0) {
#pragma unroll
    for (int j = 0; j < 8; ++j) {
      const int m = grp * 8 + j;
      const int s = s0 + m;
      float g;
      if (s == t) g = 1.f + 1e-7f;
      else        g = 1.f / (1.f + expf(-(clfacc[j] + b2v))) + 1e-7f;
      const size_t off = ((size_t)b * NT + t) * NT + s;
      outGM[off] = g;
      wsA[off]   = g;   // Jacobi working copy
    }
  }
}

// ---------------------------------------------------------------------------
// Kernel 2: parallel two-sided Jacobi eigensolver, one workgroup per batch.
// Tournament ordering: 88 disjoint (p,q) rotations per round, T-1 rounds per
// sweep. Phase 1 computes rotations, phase 2 column pass A*J and V*J,
// phase 3 row pass J^T*A. Then odd-even sort + build S replicating the
// reference's vectors * sqrt(values_desc) pairing.
// ---------------------------------------------------------------------------
__global__ __launch_bounds__(256) void jacobi_kernel(
    float* __restrict__ wsA, float* __restrict__ wsV, float* __restrict__ wsS) {
  const int b = blockIdx.x;
  const int tid = threadIdx.x;
  float* A = wsA + (size_t)b * NT * NT;
  float* V = wsV + (size_t)b * NT * NT;
  float* S = wsS + (size_t)b * NT * NT;

  __shared__ float cs[NT / 2], sn[NT / 2];
  __shared__ int   pp[NT / 2], qq[NT / 2];
  __shared__ float vals[NT];
  __shared__ int   perm[NT];

  for (int idx = tid; idx < NT * NT; idx += 256)
    V[idx] = (idx / NT == idx % NT) ? 1.f : 0.f;
  __syncthreads();

  for (int sweep = 0; sweep < NSWEEP; ++sweep) {
    for (int r = 0; r < NT - 1; ++r) {
      if (tid < NT / 2) {
        int p, q;
        if (tid == 0) { p = NT - 1; q = r % (NT - 1); }
        else {
          p = (r + tid) % (NT - 1);
          q = (r - tid + (NT - 1)) % (NT - 1);
        }
        if (p > q) { const int tmp = p; p = q; q = tmp; }
        const float app = A[(size_t)p * NT + p];
        const float aqq = A[(size_t)q * NT + q];
        const float apq = A[(size_t)p * NT + q];
        float c = 1.f, s = 0.f;
        if (fabsf(apq) > 1e-12f) {
          const float theta = 0.5f * (aqq - app) / apq;
          const float tt = (theta >= 0.f ? 1.f : -1.f) /
                           (fabsf(theta) + sqrtf(theta * theta + 1.f));
          c = 1.f / sqrtf(tt * tt + 1.f);
          s = tt * c;
        }
        cs[tid] = c; sn[tid] = s; pp[tid] = p; qq[tid] = q;
      }
      __syncthreads();
      // column pass: disjoint column pairs -> safe in parallel
      for (int idx = tid; idx < (NT / 2) * NT; idx += 256) {
        const int pr = idx / NT, row = idx % NT;
        const int p = pp[pr], q = qq[pr];
        const float c = cs[pr], s = sn[pr];
        const float ap = A[(size_t)row * NT + p], aq = A[(size_t)row * NT + q];
        A[(size_t)row * NT + p] = c * ap - s * aq;
        A[(size_t)row * NT + q] = s * ap + c * aq;
        const float vp = V[(size_t)row * NT + p], vq = V[(size_t)row * NT + q];
        V[(size_t)row * NT + p] = c * vp - s * vq;
        V[(size_t)row * NT + q] = s * vp + c * vq;
      }
      __syncthreads();
      // row pass: disjoint row pairs
      for (int idx = tid; idx < (NT / 2) * NT; idx += 256) {
        const int pr = idx / NT, col = idx % NT;
        const int p = pp[pr], q = qq[pr];
        const float c = cs[pr], s = sn[pr];
        const float ap = A[(size_t)p * NT + col], aq = A[(size_t)q * NT + col];
        A[(size_t)p * NT + col] = c * ap - s * aq;
        A[(size_t)q * NT + col] = s * ap + c * aq;
      }
      __syncthreads();
    }
  }

  // Sort eigenpairs ascending (odd-even transposition, NT passes).
  for (int g = tid; g < NT; g += 256) { vals[g] = A[(size_t)g * NT + g]; perm[g] = g; }
  __syncthreads();
  for (int it = 0; it < NT; ++it) {
    if (tid < NT / 2) {
      const int i = 2 * tid + (it & 1);
      if (i + 1 < NT && vals[i] > vals[i + 1]) {
        const float tv = vals[i]; vals[i] = vals[i + 1]; vals[i + 1] = tv;
        const int   ti = perm[i]; perm[i] = perm[i + 1]; perm[i + 1] = ti;
      }
    }
    __syncthreads();
  }

  // S[t][k] = V[t][perm[k]] * sqrt(clip(vals_asc[T-1-k]))  (reference quirk:
  // descending d is paired with ascending vectors).
  for (int idx = tid; idx < NT * NT; idx += 256) {
    const int t = idx / NT, k = idx % NT;
    float vd = vals[NT - 1 - k];
    if (vd <= 0.f) vd = 1e-7f;
    S[(size_t)t * NT + k] = V[(size_t)t * NT + perm[k]] * sqrtf(vd);
  }
}

// ---------------------------------------------------------------------------
// Kernel 3: Group = softmax((S + gumbel) / tau, axis=g). One block per (b,t).
// ---------------------------------------------------------------------------
__global__ __launch_bounds__(256) void softmax_kernel(
    const float* __restrict__ wsS, const float* __restrict__ gumbel,
    float* __restrict__ outGroup, float* __restrict__ wsGroup) {
  const int t = blockIdx.x, b = blockIdx.y;
  const int tid = threadIdx.x;
  const size_t roff = ((size_t)b * NT + t) * NT;
  __shared__ float red[256];

  float mx = -3.4e38f;
  for (int g = tid; g < NT; g += 256)
    mx = fmaxf(mx, (wsS[roff + g] + gumbel[roff + g]) * TAU_INV);
  red[tid] = mx; __syncthreads();
  for (int st = 128; st > 0; st >>= 1) {
    if (tid < st) red[tid] = fmaxf(red[tid], red[tid + st]);
    __syncthreads();
  }
  mx = red[0]; __syncthreads();

  float sum = 0.f;
  for (int g = tid; g < NT; g += 256) {
    const float e = expf((wsS[roff + g] + gumbel[roff + g]) * TAU_INV - mx);
    outGroup[roff + g] = e;
    sum += e;
  }
  red[tid] = sum; __syncthreads();
  for (int st = 128; st > 0; st >>= 1) {
    if (tid < st) red[tid] += red[tid + st];
    __syncthreads();
  }
  const float inv = 1.f / red[0];
  for (int g = tid; g < NT; g += 256) {
    const float v = outGroup[roff + g] * inv;
    outGroup[roff + g] = v;
    wsGroup[roff + g]  = v;
  }
}

// ---------------------------------------------------------------------------
// Kernel 4: per-batch is_zero flags + stable compaction positions.
// gpos[b][g] = stable index among nonzero groups, or -1 for zero groups.
// ---------------------------------------------------------------------------
__global__ __launch_bounds__(256) void flags_kernel(
    const float* __restrict__ X, const float* __restrict__ wsGroup,
    int* __restrict__ gpos) {
  const int b = blockIdx.x;
  const int tid = threadIdx.x;
  __shared__ unsigned char xnz[NT];
  __shared__ unsigned char nz[NT];
  __shared__ int pos[NT];

  for (int t = tid; t < NT; t += 256) {
    bool any = false;
    const float* xr = X + ((size_t)b * NT + t) * NROI;
    for (int i = 0; i < NROI; ++i) any = any || (xr[i] != 0.f);
    xnz[t] = any ? 1 : 0;
  }
  __syncthreads();
  for (int g = tid; g < NT; g += 256) {
    bool any = false;
    for (int t = 0; t < NT; ++t)
      any = any || ((wsGroup[((size_t)b * NT + t) * NT + g] != 0.f) && xnz[t]);
    nz[g] = any ? 1 : 0;
  }
  __syncthreads();
  if (tid == 0) {
    int c = 0;
    for (int g = 0; g < NT; ++g) pos[g] = nz[g] ? (c++) : -1;
  }
  __syncthreads();
  for (int g = tid; g < NT; g += 256) gpos[(size_t)b * NT + g] = pos[g];
}

// ---------------------------------------------------------------------------
// Kernel 5: per (b,g) FC = Zn Zn^T / (T-1), upper triangle scattered to its
// compacted output slot. Zn built in LDS as f16 (128x192 padded), 8 waves
// each own 8 of the 64 16x16 WMMA tiles, 6 k-steps of 32 (K=176 -> 192).
// ---------------------------------------------------------------------------
__global__ __launch_bounds__(256) void fc_kernel(
    const float* __restrict__ X, const float* __restrict__ wsGroup,
    const int* __restrict__ gpos, float* __restrict__ outPadded) {
  const int g = blockIdx.x, b = blockIdx.y;
  const int tid = threadIdx.x;
  __shared__ float w[NT];
  __shared__ __align__(16) _Float16 Zn[128 * 192];

  const int pos = gpos[(size_t)b * NT + g];
  if (pos < 0) return;  // uniform per block: zero groups stay zero in output

  for (int t = tid; t < NT; t += 256)
    w[t] = wsGroup[((size_t)b * NT + t) * NT + g];
  __syncthreads();

  // Per-ROI-row stats (ddof=1) and normalized f16 row.
  for (int i = tid; i < 128; i += 256) {
    if (i < NROI) {
      float s1 = 0.f, s2 = 0.f;
      for (int t = 0; t < NT; ++t) {
        const float y = w[t] * X[((size_t)b * NT + t) * NROI + i];
        s1 += y; s2 += y * y;
      }
      const float mean = s1 / (float)NT;
      float var = (s2 - s1 * s1 / (float)NT) / (float)(NT - 1);
      var = var > 0.f ? var : 0.f;
      const float inv = 1.f / (sqrtf(var) + 1e-7f);
      for (int t = 0; t < 192; ++t) {
        float z = 0.f;
        if (t < NT) {
          const float y = w[t] * X[((size_t)b * NT + t) * NROI + i];
          z = (y - mean) * inv;
        }
        Zn[i * 192 + t] = (_Float16)z;
      }
    } else {
      for (int t = 0; t < 192; ++t) Zn[i * 192 + t] = (_Float16)0.f;
    }
  }
  __syncthreads();

  const int lane = tid & 31, wave = tid >> 5;
  const int nlo = lane & 15, grp = lane >> 4;
  float* outv = outPadded + ((size_t)b * NMAXG + pos) * NPAIR;
  const float scale = 1.f / (float)(NT - 1);

  for (int tile = wave; tile < 64; tile += 8) {
    const int m0 = (tile >> 3) * 16, n0 = (tile & 7) * 16;
    v8f c = {};
#pragma unroll
    for (int ks = 0; ks < 6; ++ks) {
      const int kb = ks * 32;
      const _Float16* arow = Zn + (m0 + nlo) * 192 + kb + grp * 16;
      const v16h a = cat8(*(const v8h*)arow, *(const v8h*)(arow + 8));
      const _Float16* brow = Zn + (n0 + nlo) * 192 + kb + grp * 16;  // B[k][n]=Zn[n][k]
      const v16h bm = cat8(*(const v8h*)brow, *(const v8h*)(brow + 8));
      c = __builtin_amdgcn_wmma_f32_16x16x32_f16(false, a, false, bm,
                                                 (short)0, c, false, false);
    }
#pragma unroll
    for (int j = 0; j < 8; ++j) {
      const int i  = m0 + grp * 8 + j;  // C: M = grp*8+j
      const int jj = n0 + nlo;          // C: N = nlo
      if (i < NROI && jj < NROI && jj > i) {
        const int base = i * (NROI - 1) - (i * (i - 1)) / 2;
        outv[base + (jj - i - 1)] = c[j] * scale;
      }
    }
  }
}

// ---------------------------------------------------------------------------
extern "C" void kernel_launch(void* const* d_in, const int* in_sizes, int n_in,
                              void* d_out, int out_size, void* d_ws, size_t ws_size,
                              hipStream_t stream) {
  const float* X      = (const float*)d_in[0];
  const float* W1     = (const float*)d_in[1];
  const float* b1     = (const float*)d_in[2];
  const float* W2     = (const float*)d_in[3];
  const float* b2     = (const float*)d_in[4];
  const float* gumbel = (const float*)d_in[5];

  float* out = (float*)d_out;
  float* outPadded = out;                                      // B*MAXG*NPAIR
  float* outGM     = out + (size_t)NB * NMAXG * NPAIR;         // B*T*T
  float* outGroup  = outGM + (size_t)NB * NT * NT;             // B*T*T

  const size_t mat = (size_t)NB * NT * NT;
  float* wsA     = (float*)d_ws;
  float* wsV     = wsA + mat;
  float* wsS     = wsV + mat;
  float* wsGroup = wsS + mat;
  int*   wsGpos  = (int*)(wsGroup + mat);
  _Float16* wsW1h = (_Float16*)(wsGpos + (size_t)NB * NT);     // 16B aligned
  float* wsB1p   = (float*)(wsW1h + (size_t)NHP * 128);
  float* wsW2p   = wsB1p + NHP;

  hipMemsetAsync(outPadded, 0, (size_t)NB * NMAXG * NPAIR * sizeof(float), stream);

  prep_kernel<<<(NHP * 128 + 255) / 256, 256, 0, stream>>>(W1, b1, W2,
                                                           wsW1h, wsB1p, wsW2p);
  clf_kernel<<<dim3(NT / 16, NT, NB), 32, 0, stream>>>(X, wsW1h, wsB1p, wsW2p,
                                                       b2, outGM, wsA);
  jacobi_kernel<<<NB, 256, 0, stream>>>(wsA, wsV, wsS);
  softmax_kernel<<<dim3(NT, NB), 256, 0, stream>>>(wsS, gumbel, outGroup, wsGroup);
  flags_kernel<<<NB, 256, 0, stream>>>(X, wsGroup, wsGpos);
  fc_kernel<<<dim3(NT, NB), 256, 0, stream>>>(X, wsGroup, wsGpos, outPadded);
}
